// CharRNN_25975962206689
// MI455X (gfx1250) — matrix operand order repaired and checked by reading, not compiled, not run on hardware
//
#include <hip/hip_runtime.h>
#include <hip/hip_bf16.h>

typedef __attribute__((ext_vector_type(16))) __bf16 v16bf;
typedef __attribute__((ext_vector_type(8)))  __bf16 v8bf;
typedef __attribute__((ext_vector_type(8)))  float  v8f;

#define HDIM 2048

// CDNA5 async LDS-direct copy (ASYNCcnt-tracked), per CDNA5 ISA.
// vdst = LDS byte address (VGPR), vaddr = 64-bit global address (VGPR pair).
__device__ __forceinline__ void async_copy_b128(unsigned ldsOff, const void* g)
{
    unsigned long long ga = (unsigned long long)(uintptr_t)g;
    asm volatile("global_load_async_to_lds_b128 %0, %1, off"
                 :: "v"(ldsOff), "v"(ga) : "memory");
}
#define WAIT_ASYNC_LE4() asm volatile("s_wait_asynccnt 0x4" ::: "memory")
#define WAIT_ASYNC_0()   asm volatile("s_wait_asynccnt 0x0" ::: "memory")

// Fused GEMM:  out = act( A1 @ W1^T  [+ A2 @ W2^T]  + bias1 [+ bias2] )
//   All A operands are bf16 rows of stride HDIM; layer-1 rows come from the
//   pre-converted bf16 embedding table via a per-row token base (gather).
//   W is [N, K=HDIM] bf16 row-major: its rows ARE the WMMA B-fragments.
// Workgroup: 128 threads = 4 waves (2x2), WG tile 64x64, wave tile 32x32.
// Double-buffered LDS staged with async b128 copies, pipelined against WMMA.
__global__ __launch_bounds__(128)
void fused_gemm_wmma(
    const __bf16* __restrict__ embbf,   // [V,H] bf16 (used iff tokens != null)
    const int*    __restrict__ tokens,  // [B,T]
    int t, int T,
    const __bf16* __restrict__ a1,      // seg-0 A rows (when no gather)
    const __bf16* __restrict__ a2,      // seg-1 A rows; null => 1 segment
    const __bf16* __restrict__ w1,      // [N,HDIM] bf16
    const __bf16* __restrict__ w2,      // [N,HDIM] bf16 (seg 1)
    const float*  __restrict__ bias1,   // [N] f32
    const float*  __restrict__ bias2,   // [N] f32 or null
    __bf16*       __restrict__ outB,    // bf16 output (or null)
    float*        __restrict__ outF,    // f32 output (or null)
    int outStride, int doTanh)
{
    __shared__ __align__(16) __bf16 ldsA[2][64][32];
    __shared__ __align__(16) __bf16 ldsB[2][64][32];

    const int tid  = threadIdx.x;
    const int lane = tid & 31;
    const int wave = tid >> 5;
    const int wm   = wave >> 1;
    const int wn   = wave & 1;
    const long mBase = (long)blockIdx.y * 64;
    const long nBase = (long)blockIdx.x * 64;

    // staging role: thread -> (row r, 16-element half)
    const int r    = tid >> 1;            // 0..63
    const int kOff = (tid & 1) * 16;      // 0 or 16

    // per-segment row base pointers (computed once; inner loop is branch-free)
    const __bf16* aRowSeg[2];
    const __bf16* wRowSeg[2];
    {
        long rowA0;
        const __bf16* base0;
        if (tokens != nullptr) {
            rowA0 = (long)tokens[(mBase + r) * T + t] * HDIM;
            base0 = embbf;
        } else {
            rowA0 = (mBase + r) * HDIM;
            base0 = a1;
        }
        aRowSeg[0] = base0 + rowA0 + kOff;
        aRowSeg[1] = a2 ? a2 + (mBase + r) * HDIM + kOff : aRowSeg[0];
        wRowSeg[0] = w1 + (nBase + r) * HDIM + kOff;
        wRowSeg[1] = w2 ? w2 + (nBase + r) * HDIM + kOff : wRowSeg[0];
    }

    // issue this thread's 4 async b128 copies (32B of A, 32B of B) for a chunk
    auto stage = [&](int buf, const __bf16* aRow, const __bf16* wRow) {
        unsigned da = (unsigned)(uintptr_t)&ldsA[buf][r][kOff];
        unsigned db = (unsigned)(uintptr_t)&ldsB[buf][r][kOff];
        async_copy_b128(da,      aRow);
        async_copy_b128(da + 16, aRow + 8);
        async_copy_b128(db,      wRow);
        async_copy_b128(db + 16, wRow + 8);
    };

    v8f acc[2][2] = {};

    const int nseg  = (a2 != nullptr) ? 2 : 1;
    const int total = nseg * (HDIM / 32);      // 64 or 128 chunks

    stage(0, aRowSeg[0], wRowSeg[0]);          // prologue: chunk 0 -> buf 0
    int buf = 0;

    for (int it = 0; it < total; ++it) {
        const int nx = it + 1;
        if (nx < total) {                      // pipeline: prefetch next chunk
            const int seg = nx >> 6;           // HDIM/32 == 64 chunks per seg
            const int kc  = (nx & 63) * 32;
            stage(buf ^ 1, aRowSeg[seg] + kc, wRowSeg[seg] + kc);
            WAIT_ASYNC_LE4();                  // in-order: current buf complete
        } else {
            WAIT_ASYNC_0();
        }
        __syncthreads();

        // fragments: A and B both use contiguous k = 16*(lane/16) + s
        const int ml  = lane & 15;
        const int kbA = (lane >> 4) * 16;
        const int kbB = (lane >> 4) * 16;

        v16bf afrag[2], bfrag[2];
        #pragma unroll
        for (int i = 0; i < 2; ++i) {
            const __bf16* rp = &ldsA[buf][wm*32 + i*16 + ml][0];
            v8bf lo = *(const v8bf*)(rp + kbA);
            v8bf hi = *(const v8bf*)(rp + 8 + kbA);
            #pragma unroll
            for (int e = 0; e < 8; ++e) { afrag[i][e] = lo[e]; afrag[i][8+e] = hi[e]; }
        }
        #pragma unroll
        for (int j = 0; j < 2; ++j) {
            const __bf16* rp = &ldsB[buf][wn*32 + j*16 + ml][0];
            bfrag[j] = *(const v16bf*)(rp + kbB);
        }

        #pragma unroll
        for (int i = 0; i < 2; ++i)
            #pragma unroll
            for (int j = 0; j < 2; ++j)
                acc[i][j] = __builtin_amdgcn_wmma_f32_16x16x32_bf16(
                    false, afrag[i], false, bfrag[j],
                    (short)0, acc[i][j], false, false);

        __syncthreads();                       // buf consumed; safe to overwrite
        buf ^= 1;
    }

    // ---- epilogue: bias, tanh, stores ----
    const int nl = lane & 15;
    const int rb = (lane >> 4) * 8;
    #pragma unroll
    for (int i = 0; i < 2; ++i) {
        #pragma unroll
        for (int j = 0; j < 2; ++j) {
            long nG = nBase + wn*32 + j*16 + nl;
            float bsum = bias1 ? bias1[nG] : 0.0f;
            if (bias2) bsum += bias2[nG];
            #pragma unroll
            for (int rr = 0; rr < 8; ++rr) {
                long mG = mBase + wm*32 + i*16 + rb + rr;
                float v = acc[i][j][rr] + bsum;
                if (doTanh) v = tanhf(v);
                long o = mG * (long)outStride + nG;
                if (outB) outB[o] = (__bf16)v;
                if (outF) outF[o] = v;
            }
        }
    }
}

__global__ void cvt_f32_to_bf16(const float* __restrict__ src,
                                __bf16* __restrict__ dst, int n)
{
    int i = (blockIdx.x * blockDim.x + threadIdx.x) * 4;
    if (i + 3 < n) {
        float4 v = *(const float4*)(src + i);
        dst[i+0] = (__bf16)v.x; dst[i+1] = (__bf16)v.y;
        dst[i+2] = (__bf16)v.z; dst[i+3] = (__bf16)v.w;
    }
}

extern "C" void kernel_launch(void* const* d_in, const int* in_sizes, int n_in,
                              void* d_out, int out_size, void* d_ws, size_t ws_size,
                              hipStream_t stream)
{
    const int Vv = 512, H = 2048, L = 2, Bb = 256, Tt = 200;

    const int*   tokens = (const int*)  d_in[0];
    const float* h0     = (const float*)d_in[1];
    const float* emb    = (const float*)d_in[2];
    const float* W_ih   = (const float*)d_in[3];
    const float* W_hh   = (const float*)d_in[4];
    const float* b_ih   = (const float*)d_in[5];
    const float* b_hh   = (const float*)d_in[6];
    const float* fc_w   = (const float*)d_in[7];
    const float* fc_b   = (const float*)d_in[8];
    float* out = (float*)d_out;

    // workspace carve-up (bf16): weights+emb 36MB, states 3MB, outs 200MB
    char* p = (char*)d_ws;
    __bf16* wbih  = (__bf16*)p; p += (size_t)L * H * H * sizeof(__bf16);
    __bf16* wbhh  = (__bf16*)p; p += (size_t)L * H * H * sizeof(__bf16);
    __bf16* fcwb  = (__bf16*)p; p += (size_t)Vv * H * sizeof(__bf16);
    __bf16* embbf = (__bf16*)p; p += (size_t)Vv * H * sizeof(__bf16);
    __bf16* h1buf = (__bf16*)p; p += (size_t)2 * Bb * H * sizeof(__bf16);
    __bf16* h2in  = (__bf16*)p; p += (size_t)Bb * H * sizeof(__bf16);
    __bf16* outs  = (__bf16*)p; p += (size_t)Tt * Bb * H * sizeof(__bf16);

    // one-time (per call) f32 -> bf16 conversions, layouts preserved
    auto cvt = [&](const float* s, __bf16* d, int n) {
        cvt_f32_to_bf16<<<(n / 4 + 255) / 256, 256, 0, stream>>>(s, d, n);
    };
    cvt(W_ih, wbih, L * H * H);
    cvt(W_hh, wbhh, L * H * H);
    cvt(fc_w, fcwb, Vv * H);
    cvt(emb,  embbf, Vv * H);          // bf16 embedding table: gather is uniform
    cvt(h0,          h1buf, Bb * H);   // layer-0 init state -> ping-pong slot 0
    cvt(h0 + Bb * H, h2in,  Bb * H);   // layer-1 init state

    float* hfinal = out + (size_t)Tt * Bb * Vv;   // [L,B,H] f32 tail of d_out

    dim3 grid(H / 64, Bb / 64);   // 32 x 4 = 128 WGs per step-kernel
    for (int t = 0; t < Tt; ++t) {
        __bf16* h1r = h1buf + (size_t)(t & 1)       * Bb * H;
        __bf16* h1w = h1buf + (size_t)((t + 1) & 1) * Bb * H;
        const __bf16* h2r = (t == 0) ? h2in : outs + (size_t)(t - 1) * Bb * H;
        __bf16* h2w = outs + (size_t)t * Bb * H;
        float* f1 = (t == Tt - 1) ? hfinal : nullptr;
        float* f2 = (t == Tt - 1) ? hfinal + (size_t)Bb * H : nullptr;

        // layer 0: x_t gathered from bf16 embedding table
        fused_gemm_wmma<<<grid, 128, 0, stream>>>(
            embbf, tokens, t, Tt, nullptr, h1r,
            wbih, wbhh, b_ih, b_hh, h1w, f1, H, 1);
        // layer 1: input = layer-0 output; history written straight into outs
        fused_gemm_wmma<<<grid, 128, 0, stream>>>(
            nullptr, nullptr, 0, 0, h1w, h2r,
            wbih + (size_t)H * H, wbhh + (size_t)H * H,
            b_ih + H, b_hh + H, h2w, f2, H, 1);
    }

    // classifier: one large WMMA GEMM [T*B, H] x [H, V] -> f32 logits
    dim3 gridFC(Vv / 64, (Tt * Bb) / 64);   // 8 x 800
    fused_gemm_wmma<<<gridFC, 128, 0, stream>>>(
        nullptr, nullptr, 0, 0, outs, nullptr,
        fcwb, nullptr, fc_b, nullptr,
        nullptr, out, Vv, 0);
}
